// OCAB_26731876450679
// MI455X (gfx1250) — matrix operand order repaired and checked by reading, not compiled, not run on hardware
//
#include <hip/hip_runtime.h>
#include <hip/hip_bf16.h>
#include <math.h>

typedef __attribute__((ext_vector_type(16))) _Float16 v16h;
typedef __attribute__((ext_vector_type(8)))  float    v8f;

#define C_DIM 96
#define NHEADS 6
#define HDIM 16
#define WSZ 16
#define OWSZ 24
#define PADW 4
#define NQ 256
#define NK 576
#define IMG 128
#define NROWS 32768           /* B*H*W = 2*128*128 */
#define QK_SCALE 0.25f        /* (96/6)^-0.5 */
#define NKT (NK / 16)         /* 36 key tiles */

union F16x16 {
  v16h v;
  uint4 q[2];
  _Float16 h[16];
};

union F32x8 {
  v8f v;
  uint4 q[2];
  float f[8];
};

static __device__ __forceinline__ v8f zero8() {
  v8f z = {0.f,0.f,0.f,0.f,0.f,0.f,0.f,0.f};
  return z;
}

// ---------------------------------------------------------------------------
// f32 -> f16 convert with B-fragment panel swizzle:
//   out[(k/16)*N*16 + col*16 + (k%16)] = in[k*N + col]
// A wave32 B-fragment for k-chunk (K=32) is then 2 contiguous b128 loads.
// ---------------------------------------------------------------------------
__global__ void ocab_cvt_swz(const float* __restrict__ in, _Float16* __restrict__ out,
                             int K, int N) {
  int i = blockIdx.x * blockDim.x + threadIdx.x;
  if (i < K * N) {
    int k = i / N, c = i - k * N;
    out[((size_t)(k >> 4) * N + c) * 16 + (k & 15)] = (_Float16)in[i];
  }
}

// ---------------------------------------------------------------------------
// Relative-position bias table in WMMA C-fragment order:
//   out[((((h*16 + mt)*36 + kt)*32 + lane)*8 + r] = rpb[rpi[q,k], h]
//   with q = mt*16 + r + (lane>=16)*8, k = kt*16 + (lane&15)
// Each lane's C operand for an S-tile is then 8 contiguous floats (2 b128).
// ---------------------------------------------------------------------------
__global__ void ocab_bias(const int* __restrict__ rpi, const float* __restrict__ rpb,
                          float* __restrict__ out) {
  int i = blockIdx.x * blockDim.x + threadIdx.x;
  if (i >= NHEADS * 16 * NKT * 32 * 8) return;
  int r = i & 7;
  int l = (i >> 3) & 31;
  int kt = (i >> 8) % NKT;
  int rest = (i >> 8) / NKT;
  int mt = rest & 15;
  int h = rest >> 4;
  int q = mt * 16 + r + ((l >> 4) & 1) * 8;
  int key = kt * 16 + (l & 15);
  out[i] = rpb[rpi[q * NK + key] * NHEADS + h];
}

// ---------------------------------------------------------------------------
// LayerNorm over C=96, one wave per row, f16 output
// ---------------------------------------------------------------------------
__global__ __launch_bounds__(256) void ocab_ln(const float* __restrict__ x,
                                               const float* __restrict__ g,
                                               const float* __restrict__ b,
                                               _Float16* __restrict__ out) {
  int wid = threadIdx.x >> 5, l = threadIdx.x & 31;
  int row = blockIdx.x * 8 + wid;
  const float* xr = x + (size_t)row * C_DIM;
  float v0 = xr[l], v1 = xr[l + 32], v2 = xr[l + 64];
  float s = v0 + v1 + v2;
  float sq = v0 * v0 + v1 * v1 + v2 * v2;
#pragma unroll
  for (int m = 16; m >= 1; m >>= 1) {
    s  += __shfl_xor(s,  m, 32);
    sq += __shfl_xor(sq, m, 32);
  }
  float mean = s * (1.0f / 96.0f);
  float var  = sq * (1.0f / 96.0f) - mean * mean;
  float rstd = rsqrtf(var + 1e-5f);
  _Float16* orow = out + (size_t)row * C_DIM;
  orow[l]      = (_Float16)((v0 - mean) * rstd * g[l]      + b[l]);
  orow[l + 32] = (_Float16)((v1 - mean) * rstd * g[l + 32] + b[l + 32]);
  orow[l + 64] = (_Float16)((v2 - mean) * rstd * g[l + 64] + b[l + 64]);
}

// ---------------------------------------------------------------------------
// Generic WMMA GEMM, B pre-swizzled into panel layout.
//   C[M,N] = A[M,K](f16) @ B(f16, panels), then +bias, *scale, opt GELU,
//   opt +res(f32), store f16 and/or f32. One wave per 16x16 C tile.
// ---------------------------------------------------------------------------
__global__ __launch_bounds__(256) void ocab_gemm(
    const _Float16* __restrict__ A, const _Float16* __restrict__ Bs,
    const float* __restrict__ bias, const float* __restrict__ res,
    _Float16* __restrict__ outh, float* __restrict__ outf,
    int M, int N, int K, float scale, int gelu) {
  int wid = threadIdx.x >> 5, l = threadIdx.x & 31;
  int hi = (l >> 4) & 1, ln = l & 15;
  int ntn = N >> 4;
  int tile = blockIdx.x * 8 + wid;
  int total = (M >> 4) * ntn;
  if (tile >= total) return;
  int mt = tile / ntn, nt = tile - mt * ntn;
  int row_a = mt * 16 + ln;
  int col = nt * 16 + ln;

  v8f acc = zero8();
  for (int kb = 0; kb < K; kb += 32) {
    F16x16 au, bu;
    const _Float16* ap = A + (size_t)row_a * K + kb + hi * 16;
    au.q[0] = ((const uint4*)ap)[0];
    au.q[1] = ((const uint4*)(ap + 8))[0];
    const _Float16* bp = Bs + ((size_t)((kb >> 4) + hi) * N + col) * 16;
    bu.q[0] = ((const uint4*)bp)[0];
    bu.q[1] = ((const uint4*)bp)[1];
    acc = __builtin_amdgcn_wmma_f32_16x16x32_f16(false, au.v, false, bu.v,
                                                 (short)0, acc, false, false);
  }
  float bc = bias[col];
#pragma unroll
  for (int r = 0; r < 8; ++r) {
    int row = mt * 16 + r + hi * 8;
    float v = (acc[r] + bc) * scale;
    if (gelu) v = 0.5f * v * (1.0f + erff(v * 0.70710678118f));
    if (res)  v += res[(size_t)row * N + col];
    if (outh) outh[(size_t)row * N + col] = (_Float16)v;
    if (outf) outf[(size_t)row * N + col] = v;
  }
}

// ---------------------------------------------------------------------------
// Windowed overlapping attention: grid (128 windows, 6 heads), 256 threads.
// q pre-scaled. Bias loaded as pre-swizzled WMMA C fragments (2 b128/lane).
// Two-pass softmax with per-lane partial stats (one reduction per pass).
// k in LDS row-major; v in LDS panel-swizzled for direct B-fragment loads.
// ---------------------------------------------------------------------------
__global__ __launch_bounds__(256) void ocab_attn(
    const _Float16* __restrict__ qbuf,   // NROWS x 96 (scaled)
    const _Float16* __restrict__ kvbuf,  // NROWS x 192
    const float* __restrict__ bias_tbl,  // [6][16][36][32][8] C-frag order
    _Float16* __restrict__ attn_out) {   // NROWS x 96 (may alias qbuf: safe)
  __shared__ _Float16 k_s[NK * HDIM];           // row-major keys
  __shared__ _Float16 v_s[NK * HDIM];           // panels: [key/16][d][key%16]
  __shared__ _Float16 p_s[16 * 16 * 32];        // 16 m-tiles of 16x32 f16

  int wi = blockIdx.x, head = blockIdx.y;
  int wid = threadIdx.x >> 5, l = threadIdx.x & 31;
  int hi = (l >> 4) & 1, ln = l & 15;
  int batch = wi >> 6, win = wi & 63, wy = win >> 3, wx = win & 7;
  const float* btbl = bias_tbl + (size_t)head * NQ * NK;

  // ---- cooperative k/v gather (zero-padded); v written panel-swizzled ----
  for (int kk = threadIdx.x; kk < NK; kk += 256) {
    int ky = kk / OWSZ, kx = kk - ky * OWSZ;
    int sy = wy * WSZ - PADW + ky;
    int sx = wx * WSZ - PADW + kx;
    F16x16 kd, vd;
    kd.q[0] = kd.q[1] = vd.q[0] = vd.q[1] = (uint4){0, 0, 0, 0};
    if (sy >= 0 && sy < IMG && sx >= 0 && sx < IMG) {
      const _Float16* src =
          kvbuf + (((size_t)batch * IMG + sy) * IMG + sx) * (2 * C_DIM) + head * HDIM;
      kd.q[0] = ((const uint4*)src)[0];
      kd.q[1] = ((const uint4*)src)[1];
      const _Float16* srcv = src + C_DIM;
      vd.q[0] = ((const uint4*)srcv)[0];
      vd.q[1] = ((const uint4*)srcv)[1];
    }
    ((uint4*)&k_s[kk * HDIM])[0] = kd.q[0];
    ((uint4*)&k_s[kk * HDIM])[1] = kd.q[1];
    _Float16* vp = &v_s[(kk >> 4) * 256 + (kk & 15)];
#pragma unroll
    for (int d = 0; d < HDIM; ++d) vp[d * 16] = vd.h[d];
  }
  __syncthreads();

  // ---- per-wave state: two 16-row M-tiles ----
  float vmax[2][8], vsum[2][8];
  v8f oacc[2];
  v16h qf[2];
#pragma unroll
  for (int t = 0; t < 2; ++t) {
    int mt = wid * 2 + t;
    int qrow = mt * 16 + ln;
    int qy = qrow >> 4, qx = qrow & 15;
    int grow = (batch * IMG + wy * WSZ + qy) * IMG + (wx * WSZ + qx);
    const _Float16* qp = qbuf + (size_t)grow * C_DIM + head * HDIM;
    F16x16 a;
    a.q[0] = hi ? (uint4){0, 0, 0, 0} : ((const uint4*)qp)[0];
    a.q[1] = hi ? (uint4){0, 0, 0, 0} : ((const uint4*)qp)[1];  // K 16..31 pad (upper half-wave)
    qf[t] = a.v;
    oacc[t] = zero8();
#pragma unroll
    for (int r = 0; r < 8; ++r) { vmax[t][r] = -1e30f; vsum[t][r] = 0.f; }
  }

  // ---- pass 1: per-lane max of (q k^T + bias) ----
#pragma unroll 1
  for (int kt = 0; kt < NKT; ++kt) {
    int key0 = kt * 16;
    F16x16 bu;
    if (!hi) {
      const _Float16* kp = &k_s[(key0 + ln) * HDIM];
      bu.q[0] = ((const uint4*)kp)[0];
      bu.q[1] = ((const uint4*)kp)[1];
    } else {
      bu.q[0] = bu.q[1] = (uint4){0, 0, 0, 0};
    }
#pragma unroll
    for (int t = 0; t < 2; ++t) {
      int mt = wid * 2 + t;
      const float* cbp = btbl + (((size_t)mt * NKT + kt) * 32 + l) * 8;
      F32x8 cb;
      cb.q[0] = ((const uint4*)cbp)[0];
      cb.q[1] = ((const uint4*)cbp)[1];
      v8f sf = __builtin_amdgcn_wmma_f32_16x16x32_f16(false, qf[t], false, bu.v,
                                                      (short)0, cb.v, false, false);
#pragma unroll
      for (int r = 0; r < 8; ++r) vmax[t][r] = fmaxf(vmax[t][r], sf[r]);
    }
  }
  // one half-wave reduction per row
  float rm[2][8];
#pragma unroll
  for (int t = 0; t < 2; ++t)
#pragma unroll
    for (int r = 0; r < 8; ++r) {
      float v = vmax[t][r];
#pragma unroll
      for (int m = 8; m >= 1; m >>= 1) v = fmaxf(v, __shfl_xor(v, m, 16));
      rm[t][r] = v;
    }

  // ---- pass 2: recompute S, exp, per-lane sum, P@V accumulate ----
#pragma unroll 1
  for (int kc = 0; kc < NK / 32; ++kc) {
#pragma unroll
    for (int sub = 0; sub < 2; ++sub) {
      int kt = kc * 2 + sub;
      int key0 = kt * 16;
      F16x16 bu;
      if (!hi) {
        const _Float16* kp = &k_s[(key0 + ln) * HDIM];
        bu.q[0] = ((const uint4*)kp)[0];
        bu.q[1] = ((const uint4*)kp)[1];
      } else {
        bu.q[0] = bu.q[1] = (uint4){0, 0, 0, 0};
      }
#pragma unroll
      for (int t = 0; t < 2; ++t) {
        int mt = wid * 2 + t;
        const float* cbp = btbl + (((size_t)mt * NKT + kt) * 32 + l) * 8;
        F32x8 cb;
        cb.q[0] = ((const uint4*)cbp)[0];
        cb.q[1] = ((const uint4*)cbp)[1];
        v8f sf = __builtin_amdgcn_wmma_f32_16x16x32_f16(false, qf[t], false, bu.v,
                                                        (short)0, cb.v, false, false);
        _Float16* pp = &p_s[mt * 512];
#pragma unroll
        for (int r = 0; r < 8; ++r) {
          float p = __expf(sf[r] - rm[t][r]);
          vsum[t][r] += p;
          pp[(r + hi * 8) * 32 + sub * 16 + ln] = (_Float16)p;
        }
      }
    }
    // same-wave LDS RAW: P stores -> A-fragment reload
    asm volatile("s_wait_dscnt 0" ::: "memory");

    // B = V chunk (32 keys x 16 dims), panel-swizzled: 2 contiguous b128 loads
    F16x16 vu;
    const _Float16* vp = &v_s[((kc * 2 + hi) * 16 + ln) * 16];
    vu.q[0] = ((const uint4*)vp)[0];
    vu.q[1] = ((const uint4*)vp)[1];
#pragma unroll
    for (int t = 0; t < 2; ++t) {
      int mt = wid * 2 + t;
      const _Float16* ap = &p_s[mt * 512] + ln * 32 + hi * 16;
      F16x16 au;
      au.q[0] = ((const uint4*)ap)[0];
      au.q[1] = ((const uint4*)(ap + 8))[0];
      oacc[t] = __builtin_amdgcn_wmma_f32_16x16x32_f16(false, au.v, false, vu.v,
                                                       (short)0, oacc[t], false, false);
    }
  }

  // ---- reduce row sums once, normalize, scatter to (b, h*w, c) ----
#pragma unroll
  for (int t = 0; t < 2; ++t) {
    int mt = wid * 2 + t;
#pragma unroll
    for (int r = 0; r < 8; ++r) {
      float s = vsum[t][r];
#pragma unroll
      for (int m = 8; m >= 1; m >>= 1) s += __shfl_xor(s, m, 16);
      int qr = mt * 16 + r + hi * 8;
      int qy = qr >> 4, qx = qr & 15;
      int grow = (batch * IMG + wy * WSZ + qy) * IMG + (wx * WSZ + qx);
      attn_out[(size_t)grow * C_DIM + head * HDIM + ln] =
          (_Float16)(oacc[t][r] / s);
    }
  }
}

// ---------------------------------------------------------------------------
// Host-side launch sequence
// ---------------------------------------------------------------------------
extern "C" void kernel_launch(void* const* d_in, const int* in_sizes, int n_in,
                              void* d_out, int out_size, void* d_ws, size_t ws_size,
                              hipStream_t stream) {
  (void)in_sizes; (void)n_in; (void)out_size; (void)ws_size;
  const float* x      = (const float*)d_in[0];
  const int*   rpi    = (const int*)d_in[1];
  const float* n1g    = (const float*)d_in[4];
  const float* n1b    = (const float*)d_in[5];
  const float* q_w    = (const float*)d_in[6];
  const float* q_b    = (const float*)d_in[7];
  const float* kv_w   = (const float*)d_in[8];
  const float* kv_b   = (const float*)d_in[9];
  const float* rpb    = (const float*)d_in[10];
  const float* proj_w = (const float*)d_in[11];
  const float* proj_b = (const float*)d_in[12];
  const float* n2g    = (const float*)d_in[13];
  const float* n2b    = (const float*)d_in[14];
  const float* w1     = (const float*)d_in[15];
  const float* b1     = (const float*)d_in[16];
  const float* w2     = (const float*)d_in[17];
  const float* b2     = (const float*)d_in[18];

  char* ws = (char*)d_ws;
  size_t off = 0;
  auto alloc = [&](size_t bytes) -> void* {
    void* p = ws + off;
    off += (bytes + 255) & ~(size_t)255;
    return p;
  };
  _Float16* xn   = (_Float16*)alloc((size_t)NROWS * C_DIM * 2);     // ln1 out, reused as ln2 out
  _Float16* qbuf = (_Float16*)alloc((size_t)NROWS * C_DIM * 2);     // q, then attn_out (alias-safe)
  _Float16* kvb  = (_Float16*)alloc((size_t)NROWS * 2 * C_DIM * 2); // kv, then gelu hidden
  float*    x2   = (float*)alloc((size_t)NROWS * C_DIM * 4);
  float*    btbl = (float*)alloc((size_t)NHEADS * NQ * NK * 4);
  _Float16* wqh  = (_Float16*)alloc((size_t)C_DIM * C_DIM * 2);
  _Float16* wkvh = (_Float16*)alloc((size_t)C_DIM * 2 * C_DIM * 2);
  _Float16* wph  = (_Float16*)alloc((size_t)C_DIM * C_DIM * 2);
  _Float16* w1h  = (_Float16*)alloc((size_t)C_DIM * 2 * C_DIM * 2);
  _Float16* w2h  = (_Float16*)alloc((size_t)2 * C_DIM * C_DIM * 2);

  ocab_cvt_swz<<<(96 * 96 + 255) / 256, 256, 0, stream>>>(q_w, wqh, 96, 96);
  ocab_cvt_swz<<<(96 * 192 + 255) / 256, 256, 0, stream>>>(kv_w, wkvh, 96, 192);
  ocab_cvt_swz<<<(96 * 96 + 255) / 256, 256, 0, stream>>>(proj_w, wph, 96, 96);
  ocab_cvt_swz<<<(96 * 192 + 255) / 256, 256, 0, stream>>>(w1, w1h, 96, 192);
  ocab_cvt_swz<<<(192 * 96 + 255) / 256, 256, 0, stream>>>(w2, w2h, 192, 96);
  ocab_bias<<<(NHEADS * NQ * NK + 255) / 256, 256, 0, stream>>>(rpi, rpb, btbl);

  // LN1
  ocab_ln<<<NROWS / 8, 256, 0, stream>>>(x, n1g, n1b, xn);
  // q = xn @ q_w + q_b, pre-scaled by 1/sqrt(d)
  ocab_gemm<<<(2048 * 6) / 8, 256, 0, stream>>>(xn, wqh, q_b, nullptr, qbuf, nullptr,
                                                NROWS, 96, 96, QK_SCALE, 0);
  // kv = xn @ kv_w + kv_b
  ocab_gemm<<<(2048 * 12) / 8, 256, 0, stream>>>(xn, wkvh, kv_b, nullptr, kvb, nullptr,
                                                 NROWS, 192, 96, 1.0f, 0);
  // attention (128 windows x 6 heads)
  ocab_attn<<<dim3(128, 6), 256, 0, stream>>>(qbuf, kvb, btbl, qbuf);
  // x2 = attn @ proj_w + proj_b + shortcut(x)   (f32)
  ocab_gemm<<<(2048 * 6) / 8, 256, 0, stream>>>(qbuf, wph, proj_b, x, nullptr, x2,
                                                NROWS, 96, 96, 1.0f, 0);
  // LN2
  ocab_ln<<<NROWS / 8, 256, 0, stream>>>(x2, n2g, n2b, xn);
  // hidden = gelu(y @ w1 + b1)
  ocab_gemm<<<(2048 * 12) / 8, 256, 0, stream>>>(xn, w1h, b1, nullptr, kvb, nullptr,
                                                 NROWS, 192, 96, 1.0f, 1);
  // out = hidden @ w2 + b2 + x2
  ocab_gemm<<<(2048 * 6) / 8, 256, 0, stream>>>(kvb, w2h, b2, x2, nullptr, (float*)d_out,
                                                NROWS, 96, 192, 1.0f, 0);
}
